// Hessian2_16501264351426
// MI455X (gfx1250) — matrix operand order repaired and checked by reading, not compiled, not run on hardware
//
#include <hip/hip_runtime.h>
#include <hip/hip_bf16.h>
#include <stdint.h>

// ---------------------------------------------------------------------------
// MI455X (gfx1250) fused Hessian-network kernel.
//   * all heavy GEMMs run as v_wmma_f32_16x16x32_bf16 (f32 accumulate)
//   * weights pre-packed to bf16 / transposed in d_ws (prep kernel)
//   * per-block weight staging into LDS via async-to-LDS (guarded builtins)
//   * symmetrize+segment_sum fused into f32 atomics (output stays in L2)
// ---------------------------------------------------------------------------

typedef __attribute__((ext_vector_type(16))) __bf16 v16bf;
typedef __attribute__((ext_vector_type(8)))  __bf16 v8bf;
typedef __attribute__((ext_vector_type(8)))  float  v8f;
typedef __attribute__((__vector_size__(4 * sizeof(int)))) int v4i_t;

#define N_ATOMS_C 200000
#define N_MOL_C   8000
#define WAVES_PB  4
#define ATOMS_PB  64
#define NBLOCKS_C (N_ATOMS_C / ATOMS_PB)   // 3125 (exact)

// ---- d_ws layout (bytes) ---------------------------------------------------
#define WS_WMIXT   0        // bf16 [128][128]  geb0.Wmix^T (N-major)
#define WS_D1WT    32768    // bf16 [64][192]   geb0.d1.W^T
#define WS_D2WT    57344    // bf16 [128][64]   geb0.d2.W^T
#define WS_W2T     73728    // bf16 [736][96]   concat fnn {p,s,v} d2.W^T (padded)
#define WS_B2CAT   215040   // f32  [736]       bp2+bs2+bv2 (padded)
#define WS_TOTAL   217984

// ---- LDS layout ------------------------------------------------------------
#define LDS_WEIGHTS 73728                 // WmixT + D1WT + D2WT staged in LDS
#define AR_VBUF 0                         // bf16 [48][128]  v tile (comp-major)
#define AR_VMIX 12288                     // bf16 [48][128]  vmix result
#define AR_CTX  24576                     // bf16 [16][192]  [s | Vn]
#define AR_H    30720                     // bf16 [16][64]   silu(d1)
#define AR_X    32768                     // f32  [16][128]  d2 out
#define AR_HCAT 40960                     // bf16 [16][96]   [hp|hs|hv] padded
#define AR_BYTES 44032
#define SMEM_BYTES (LDS_WEIGHTS + WAVES_PB * AR_BYTES)   // 249856 <= 320KB/WGP

// Optional gfx1250 async-to-LDS path (probe-guarded; falls back to ds stores).
#if defined(__has_builtin)
#if __has_builtin(__builtin_amdgcn_global_load_async_to_lds_b128) && \
    __has_builtin(__builtin_amdgcn_s_wait_asynccnt)
#define USE_ASYNC_LDS 1
#endif
#endif

// ---- helpers ---------------------------------------------------------------
__device__ __host__ inline __bf16 f2bf(float f) {
  union { float f; uint32_t u; } c; c.f = f;
  uint32_t r = c.u + 0x7FFFu + ((c.u >> 16) & 1u);
  uint16_t h = (uint16_t)(r >> 16);
  __bf16 o; __builtin_memcpy(&o, &h, 2);
  return o;
}
__device__ __host__ inline float bf2f(__bf16 b) {
  uint16_t h; __builtin_memcpy(&h, &b, 2);
  union { uint32_t u; float f; } c; c.u = ((uint32_t)h) << 16;
  return c.f;
}
__device__ inline float silu_f(float x) { return x / (1.f + __expf(-x)); }

// A fragment (16xK activation tile, row-major, row = lane&15).
// A layout: lanes 0-15 hold K[0..15], lanes 16-31 K[16..31] (matches load_b).
__device__ inline v16bf load_a(const __bf16* base, int pitch, int kbase) {
  const int l = threadIdx.x & 31;
  const __bf16* p = base + (l & 15) * pitch + kbase + ((l & 16) ? 16 : 0);
  v8bf lo = *(const v8bf*)(p);
  v8bf hi = *(const v8bf*)(p + 8);
  return __builtin_shufflevector(lo, hi, 0,1,2,3,4,5,6,7,8,9,10,11,12,13,14,15);
}
// B fragment (Kx16 weight tile stored transposed N-major [N][Kpitch]).
// B layout used here: lane n = column, 16 contiguous K values (low/high half by lane>=16).
__device__ inline v16bf load_b(const __bf16* base, int pitch, int kbase) {
  const int l = threadIdx.x & 31;
  const __bf16* p = base + (l & 15) * pitch + kbase + ((l & 16) ? 16 : 0);
  return *(const v16bf*)(p);
}
__device__ inline v8f wmma_bf16(v16bf a, v16bf b, v8f c) {
  return __builtin_amdgcn_wmma_f32_16x16x32_bf16(false, a, false, b, (short)0, c,
                                                 false, false);
}
// D layout: VGPR i, lanes 0-15 -> (M=i, N=lane); lanes 16-31 -> (M=8+i, N=lane-16)
__device__ inline void store_d_bf16(__bf16* base, int pitch, int nbase, const v8f& c) {
  const int l = threadIdx.x & 31;
  const int n = nbase + (l & 15);
  const int m0 = (l & 16) ? 8 : 0;
#pragma unroll
  for (int i = 0; i < 8; ++i) base[(m0 + i) * pitch + n] = f2bf(c[i]);
}

// ---- kernel 0: zero output (atomics accumulate into it) --------------------
__global__ void zero_out_kernel(float* __restrict__ o, int n) {
  int i = blockIdx.x * blockDim.x + threadIdx.x;
  if (i < n) o[i] = 0.f;
}

// ---- kernel 1: pack/transpose weights to bf16 in d_ws ----------------------
__global__ void prep_weights_kernel(const float* __restrict__ wmix0,
                                    const float* __restrict__ d1w0,
                                    const float* __restrict__ d2w0,
                                    const float* __restrict__ wp2,
                                    const float* __restrict__ ws2,
                                    const float* __restrict__ wv2,
                                    const float* __restrict__ bp2,
                                    const float* __restrict__ bs2,
                                    const float* __restrict__ bv2,
                                    char* __restrict__ wsbuf) {
  __bf16* wmixT = (__bf16*)(wsbuf + WS_WMIXT);
  __bf16* d1T   = (__bf16*)(wsbuf + WS_D1WT);
  __bf16* d2T   = (__bf16*)(wsbuf + WS_D2WT);
  __bf16* w2T   = (__bf16*)(wsbuf + WS_W2T);
  float*  b2c   = (float*)(wsbuf + WS_B2CAT);
  for (int idx = blockIdx.x * blockDim.x + threadIdx.x; idx < 108256;
       idx += gridDim.x * blockDim.x) {
    if (idx < 16384) {                       // geb0.Wmix (128,128) -> [n][k]
      int n = idx >> 7, k = idx & 127;
      wmixT[n * 128 + k] = f2bf(wmix0[k * 128 + n]);
    } else if (idx < 28672) {                // geb0.d1.W (192,64) -> [n][k]
      int t = idx - 16384; int n = t / 192, k = t - n * 192;
      d1T[n * 192 + k] = f2bf(d1w0[k * 64 + n]);
    } else if (idx < 36864) {                // geb0.d2.W (64,128) -> [n][k]
      int t = idx - 28672; int n = t >> 6, k = t & 63;
      d2T[n * 64 + k] = f2bf(d2w0[k * 128 + n]);
    } else if (idx < 107520) {               // concat fnn d2.W (30,729)x3 -> [736][96]
      int t = idx - 36864; int n = t / 96, k = t - n * 96;
      int b = k >> 5, kk = k & 31;
      float v = 0.f;
      if (kk < 30 && n < 729) {
        const float* w = (b == 0) ? wp2 : (b == 1) ? ws2 : wv2;
        v = w[kk * 729 + n];
      }
      w2T[n * 96 + k] = f2bf(v);
    } else {                                 // summed bias (729, padded to 736)
      int n = idx - 107520;
      b2c[n] = (n < 729) ? (bp2[n] + bs2[n] + bv2[n]) : 0.f;
    }
  }
}

// ---- kernel 2: fused network ----------------------------------------------
__global__ void __launch_bounds__(128, 1)
hess_fused_kernel(const float* __restrict__ pos,
                  const float* __restrict__ srep,
                  const float* __restrict__ vrep,
                  const int*   __restrict__ idxm,
                  const float* __restrict__ b1,     // geb0.d1.b (64)
                  const float* __restrict__ b2,     // geb0.d2.b (128)
                  const float* __restrict__ wmix1,  // geb1.Wmix (64,2)
                  const float* __restrict__ d1w1,   // geb1.d1.W (65)
                  const float* __restrict__ d1b1,   // geb1.d1.b (1)
                  const float* __restrict__ d2w1,   // geb1.d2.W (2)
                  const float* __restrict__ d2b1,   // geb1.d2.b (2)
                  const float* __restrict__ wp1, const float* __restrict__ bp1,
                  const float* __restrict__ ws1, const float* __restrict__ bs1,
                  const float* __restrict__ wv1, const float* __restrict__ bv1,
                  const char*  __restrict__ wsbuf,
                  float* __restrict__ dout) {
  extern __shared__ float4 smem_f4[];
  char* smem = (char*)smem_f4;
  const int lane = threadIdx.x & 31;
  const int wave = threadIdx.x >> 5;
  const int atom0 = (blockIdx.x * WAVES_PB + wave) * 16;

  // Pull this tile's streamed inputs toward the caches early (global_prefetch_b8).
  __builtin_prefetch(vrep + (size_t)atom0 * 3 * 128, 0, 3);
  __builtin_prefetch(srep + (size_t)atom0 * 128, 0, 3);

  // --- stage 0: block-cooperative weight staging into LDS (72KB) -----------
  {
    const int n16 = LDS_WEIGHTS / 16;        // 4608 16-byte chunks
#ifdef USE_ASYNC_LDS
    for (int i = threadIdx.x; i < n16; i += blockDim.x) {
      __builtin_amdgcn_global_load_async_to_lds_b128(
          (__attribute__((address_space(1))) v4i_t*)(uintptr_t)(wsbuf + i * 16),
          (__attribute__((address_space(3))) v4i_t*)(uintptr_t)(smem + i * 16),
          0, 0);
    }
    __builtin_amdgcn_s_wait_asynccnt(0);
#else
    const uint4* g = (const uint4*)wsbuf;
    uint4* s = (uint4*)smem;
    for (int i = threadIdx.x; i < n16; i += blockDim.x) s[i] = g[i];
#endif
    __syncthreads();
  }
  const __bf16* sWmixT = (const __bf16*)(smem + WS_WMIXT);
  const __bf16* sD1WT  = (const __bf16*)(smem + WS_D1WT);
  const __bf16* sD2WT  = (const __bf16*)(smem + WS_D2WT);
  const __bf16* gW2T   = (const __bf16*)(wsbuf + WS_W2T);    // streamed from L2
  const float*  gB2c   = (const float*)(wsbuf + WS_B2CAT);

  char* arena = smem + LDS_WEIGHTS + wave * AR_BYTES;
  __bf16* vbuf = (__bf16*)(arena + AR_VBUF);
  __bf16* vmix = (__bf16*)(arena + AR_VMIX);
  __bf16* ctx  = (__bf16*)(arena + AR_CTX);
  __bf16* hbuf = (__bf16*)(arena + AR_H);
  float*  xbuf = (float*)(arena + AR_X);
  __bf16* hcat = (__bf16*)(arena + AR_HCAT);

  // --- stage 1: load tile inputs, f32 -> bf16 ------------------------------
  {
    const float4* v4 = (const float4*)vrep;
    for (int idx = lane; idx < 48 * 32; idx += 32) {   // rows: r = comp*16 + atom
      int r = idx >> 5, q = idx & 31;
      int a = r & 15, comp = r >> 4;
      float4 f = v4[((size_t)(atom0 + a) * 3 + comp) * 32 + q];
      __bf16* d = vbuf + r * 128 + q * 4;
      d[0] = f2bf(f.x); d[1] = f2bf(f.y); d[2] = f2bf(f.z); d[3] = f2bf(f.w);
    }
    const float4* s4 = (const float4*)srep;
    for (int idx = lane; idx < 16 * 32; idx += 32) {
      int a = idx >> 5, q = idx & 31;
      float4 f = s4[(size_t)(atom0 + a) * 32 + q];
      __bf16* d = ctx + a * 192 + q * 4;
      d[0] = f2bf(f.x); d[1] = f2bf(f.y); d[2] = f2bf(f.z); d[3] = f2bf(f.w);
    }
  }

  // --- stage 2: vmix = v @ Wmix  (48x128x128, 96 WMMA) ---------------------
  for (int mb = 0; mb < 3; ++mb) {
    for (int nt = 0; nt < 8; ++nt) {
      v8f c = {};
#pragma unroll
      for (int kb = 0; kb < 4; ++kb) {
        v16bf a = load_a(vbuf + mb * 16 * 128, 128, kb * 32);
        v16bf b = load_b(sWmixT + nt * 16 * 128, 128, kb * 32);
        c = wmma_bf16(a, b, c);
      }
      store_d_bf16(vmix + mb * 16 * 128, 128, nt * 16, c);
    }
  }

  // --- stage 3: vec_Vn = ||vec_V||_c  -> ctx[:,128:192] --------------------
  for (int idx = lane; idx < 16 * 64; idx += 32) {
    int a = idx >> 6, o = idx & 63;
    float acc = 0.f;
#pragma unroll
    for (int comp = 0; comp < 3; ++comp) {
      float x = bf2f(vmix[(comp * 16 + a) * 128 + o]);
      acc += x * x;
    }
    ctx[a * 192 + 128 + o] = f2bf(sqrtf(acc));
  }

  // --- stage 4: h = silu(ctx @ d1.W + b1)  (16x192x64, 24 WMMA) ------------
  for (int nt = 0; nt < 4; ++nt) {
    v8f c = {};
#pragma unroll
    for (int kb = 0; kb < 6; ++kb) {
      v16bf a = load_a(ctx, 192, kb * 32);
      v16bf b = load_b(sD1WT + nt * 16 * 192, 192, kb * 32);
      c = wmma_bf16(a, b, c);
    }
    const int n = nt * 16 + (lane & 15);
    const float bias = b1[n];
    const int m0 = (lane & 16) ? 8 : 0;
#pragma unroll
    for (int i = 0; i < 8; ++i)
      hbuf[(m0 + i) * 64 + n] = f2bf(silu_f(c[i] + bias));
  }

  // --- stage 5: x = h @ d2.W + b2  (16x64x128, 16 WMMA), keep f32 ----------
  for (int nt = 0; nt < 8; ++nt) {
    v8f c = {};
#pragma unroll
    for (int kb = 0; kb < 2; ++kb) {
      v16bf a = load_a(hbuf, 64, kb * 32);
      v16bf b = load_b(sD2WT + nt * 16 * 64, 64, kb * 32);
      c = wmma_bf16(a, b, c);
    }
    const int n = nt * 16 + (lane & 15);
    const float bias = b2[n];
    const int m0 = (lane & 16) ? 8 : 0;
#pragma unroll
    for (int i = 0; i < 8; ++i) xbuf[(m0 + i) * 128 + n] = c[i] + bias;
  }

  // --- stage 6: geb1 (N=1/2, VALU) + fnn hidden layers -> hcat -------------
  if (lane < 16) {
    const int a = lane;
    const int atom = atom0 + a;
    const float* xr = xbuf + a * 128;
    float dot1 = d1b1[0];
    float vm00 = 0, vm01 = 0, vm10 = 0, vm11 = 0, vm20 = 0, vm21 = 0;
    for (int o = 0; o < 64; ++o) {
      float so = silu_f(xr[o]);               // s_out (silu)
      dot1 += so * d1w1[o];
      float g  = xr[64 + o];                  // gate
      float w0 = wmix1[o * 2 + 0], w1 = wmix1[o * 2 + 1];
      float vw0 = g * bf2f(vmix[(0 * 16 + a) * 128 + 64 + o]);  // v_out = gate*vec_W
      float vw1 = g * bf2f(vmix[(1 * 16 + a) * 128 + 64 + o]);
      float vw2 = g * bf2f(vmix[(2 * 16 + a) * 128 + 64 + o]);
      vm00 += vw0 * w0; vm01 += vw0 * w1;
      vm10 += vw1 * w0; vm11 += vw1 * w1;
      vm20 += vw2 * w0; vm21 += vw2 * w1;
    }
    float vn1 = sqrtf(vm00 * vm00 + vm10 * vm10 + vm20 * vm20);
    dot1 += vn1 * d1w1[64];
    float h1  = silu_f(dot1);
    float l0v = h1 * d2w1[0] + d2b1[0];       // geb1 s_out (no activation)
    float g1  = h1 * d2w1[1] + d2b1[1];       // geb1 gate
    float l1x = g1 * vm01, l1y = g1 * vm11, l1z = g1 * vm21;
    float p0 = pos[(size_t)atom * 3 + 0];
    float p1 = pos[(size_t)atom * 3 + 1];
    float p2 = pos[(size_t)atom * 3 + 2];
    __bf16* hr = hcat + a * 96;
#pragma unroll
    for (int t = 0; t < 32; ++t) {
      if (t < 30) {
        float hp = silu_f(p0 * wp1[t] + p1 * wp1[30 + t] + p2 * wp1[60 + t] + bp1[t]);
        float hs = silu_f(l0v * ws1[t] + bs1[t]);
        float hv = silu_f(l1x * wv1[t] + l1y * wv1[30 + t] + l1z * wv1[60 + t] + bv1[t]);
        hr[t] = f2bf(hp); hr[32 + t] = f2bf(hs); hr[64 + t] = f2bf(hv);
      } else {
        __bf16 z = f2bf(0.f);
        hr[t] = z; hr[32 + t] = z; hr[64 + t] = z;
      }
    }
  }

  // --- stage 7: hess = hcat @ W2cat + bias, fused sym + segment_sum --------
  // (H + H^T - diag(H)*I): element (i,j) adds to out[i][j] and (i!=j) out[j][i].
  const int m0 = (lane & 16) ? 8 : 0;
  int mol[8]; bool same = true;
#pragma unroll
  for (int r = 0; r < 8; ++r) {
    mol[r] = idxm[atom0 + m0 + r];
    if (mol[r] != mol[0]) same = false;
  }
  for (int nt = 0; nt < 46; ++nt) {
    v8f c = {};
#pragma unroll
    for (int kb = 0; kb < 3; ++kb) {
      v16bf a = load_a(hcat, 96, kb * 32);
      v16bf b = load_b(gW2T + nt * 16 * 96, 96, kb * 32);
      c = wmma_bf16(a, b, c);
    }
    const int n = nt * 16 + (lane & 15);
    if (n < 729) {
      const float bias = gB2c[n];
      const int i = n / 27, j = n - i * 27;
      if (same) {                       // sorted idx_m: usually one molecule/half-tile
        float sacc = 8.f * bias;
#pragma unroll
        for (int r = 0; r < 8; ++r) sacc += c[r];
        float* o = dout + (size_t)mol[0] * 729;
        atomicAdd(o + i * 27 + j, sacc);
        if (i != j) atomicAdd(o + j * 27 + i, sacc);
      } else {
#pragma unroll
        for (int r = 0; r < 8; ++r) {
          float v = c[r] + bias;
          float* o = dout + (size_t)mol[r] * 729;
          atomicAdd(o + i * 27 + j, v);
          if (i != j) atomicAdd(o + j * 27 + i, v);
        }
      }
    }
  }
}

// ---------------------------------------------------------------------------
// d_in layout (setup_inputs() dict insertion order, params flattened in
// declaration order): 0 positions, 1 scalar_rep, 2 vector_rep, 3 idx_m,
// 4 geb0.Wmix, 5 geb0.d1.W, 6 geb0.d1.b, 7 geb0.d2.W, 8 geb0.d2.b,
// 9 geb1.Wmix, 10 geb1.d1.W, 11 geb1.d1.b, 12 geb1.d2.W, 13 geb1.d2.b,
// 14 fnn_p.d1.W, 15 fnn_p.d1.b, 16 fnn_p.d2.W, 17 fnn_p.d2.b,
// 18 fnn_s.d1.W, 19 fnn_s.d1.b, 20 fnn_s.d2.W, 21 fnn_s.d2.b,
// 22 fnn_v.d1.W, 23 fnn_v.d1.b, 24 fnn_v.d2.W, 25 fnn_v.d2.b
// ---------------------------------------------------------------------------
extern "C" void kernel_launch(void* const* d_in, const int* in_sizes, int n_in,
                              void* d_out, int out_size, void* d_ws, size_t ws_size,
                              hipStream_t stream) {
  const float* pos  = (const float*)d_in[0];
  const float* srep = (const float*)d_in[1];
  const float* vrep = (const float*)d_in[2];
  const int*   idxm = (const int*)d_in[3];
  const float* wmix0 = (const float*)d_in[4];
  const float* d1w0  = (const float*)d_in[5];
  const float* b1    = (const float*)d_in[6];
  const float* d2w0  = (const float*)d_in[7];
  const float* b2    = (const float*)d_in[8];
  const float* wmix1 = (const float*)d_in[9];
  const float* d1w1  = (const float*)d_in[10];
  const float* d1b1  = (const float*)d_in[11];
  const float* d2w1  = (const float*)d_in[12];
  const float* d2b1  = (const float*)d_in[13];
  const float* wp1 = (const float*)d_in[14];
  const float* bp1 = (const float*)d_in[15];
  const float* wp2 = (const float*)d_in[16];
  const float* bp2 = (const float*)d_in[17];
  const float* ws1 = (const float*)d_in[18];
  const float* bs1 = (const float*)d_in[19];
  const float* ws2 = (const float*)d_in[20];
  const float* bs2 = (const float*)d_in[21];
  const float* wv1 = (const float*)d_in[22];
  const float* bv1 = (const float*)d_in[23];
  const float* wv2 = (const float*)d_in[24];
  const float* bv2 = (const float*)d_in[25];
  float* out = (float*)d_out;
  char*  ws  = (char*)d_ws;
  (void)in_sizes; (void)n_in; (void)ws_size;

  // Output is accumulated via atomics -> zero it every call (deterministic).
  zero_out_kernel<<<(out_size + 255) / 256, 256, 0, stream>>>(out, out_size);

  prep_weights_kernel<<<424, 256, 0, stream>>>(wmix0, d1w0, d2w0, wp2, ws2, wv2,
                                               bp2, bs2, bv2, ws);

  static int smem_attr_set = 0;  // host-side only; does not affect device work
  if (!smem_attr_set) {
    (void)hipFuncSetAttribute((const void*)hess_fused_kernel,
                              hipFuncAttributeMaxDynamicSharedMemorySize,
                              SMEM_BYTES);
    smem_attr_set = 1;
  }
  hess_fused_kernel<<<NBLOCKS_C, WAVES_PB * 32, SMEM_BYTES, stream>>>(
      pos, srep, vrep, idxm, b1, b2, wmix1, d1w1, d1b1, d2w1, d2b1,
      wp1, bp1, ws1, bs1, wv1, bv1, ws, out);
}
